// MoEBlock_26276609917522
// MI455X (gfx1250) — compile-verified, matrix operand order checked by reading, not run on hardware
//
#include <hip/hip_runtime.h>
#include <hip/hip_bf16.h>
#include <math.h>

// ---------------------------------------------------------------------------
// MoE block for MI455X (gfx1250): bf16 WMMA (v_wmma_f32_16x16x32_bf16),
// wave32, top-2 sparse expert dispatch, M=32 token tiles so every weight
// (B) fragment feeds two WMMAs -> halved weight traffic vs M=16.
// ---------------------------------------------------------------------------

typedef __attribute__((ext_vector_type(16))) __bf16 v16bf;
typedef __attribute__((ext_vector_type(8)))  float  v8f;
typedef unsigned short u16;

#define BB 4
#define NN 2048
#define DD 1024
#define EE 8
#define FF 4096
#define TT (BB * NN)   // 8192 tokens

union Frag { v16bf v; uint4 u[2]; };

__device__ __forceinline__ u16 f2bf(float f) {
  unsigned u = __float_as_uint(f);
  unsigned r = (u + 0x7FFFu + ((u >> 16) & 1u)) >> 16;  // round-to-nearest-even
  return (u16)r;
}

__device__ __forceinline__ float wredsum(float s) {
  #pragma unroll
  for (int off = 16; off > 0; off >>= 1) s += __shfl_xor(s, off, 32);
  return s;
}

__device__ __forceinline__ float gelu_tanh(float v) {
  return 0.5f * v * (1.0f + tanhf(0.7978845608028654f * (v + 0.044715f * v * v * v)));
}

// ---------------------------------------------------------------------------
// Kernel 0: zero the per-expert token counters
// ---------------------------------------------------------------------------
__global__ void zero_counts(int* cnt) {
  if (threadIdx.x < EE) cnt[threadIdx.x] = 0;
}

// ---------------------------------------------------------------------------
// Kernel 1: fp32 -> bf16 conversion with transpose.
//   src: [E][R][C] fp32, dst: [E][C][R] bf16
// ---------------------------------------------------------------------------
__global__ void convert_transpose(const float* __restrict__ src,
                                  u16* __restrict__ dst,
                                  int R, int C, long long total) {
  long long stride = (long long)gridDim.x * blockDim.x;
  for (long long o = (long long)blockIdx.x * blockDim.x + threadIdx.x;
       o < total; o += stride) {
    long long rc  = (long long)R * C;
    int e  = (int)(o / rc);
    long long rem = o - (long long)e * rc;
    int c = (int)(rem / R);
    int r = (int)(rem - (long long)c * R);
    float v = src[((long long)e * R + r) * C + c];
    dst[o] = f2bf(v);
  }
}

// ---------------------------------------------------------------------------
// Kernel 2: LayerNorm + top-2 gate + residual init + expert compaction.
// One wave32 per token. Lane l owns d = l + 32k, k = 0..31.
// ---------------------------------------------------------------------------
__global__ void ln_gate(const float* __restrict__ x,
                        const float* __restrict__ gamma,
                        const float* __restrict__ beta,
                        const float* __restrict__ Wg,
                        float* __restrict__ out,        // residual init
                        u16*  __restrict__ xnb,         // [T][D] bf16
                        int*  __restrict__ cnt,         // [E]
                        int*  __restrict__ tokL,        // [E][T]
                        float* __restrict__ gateL) {    // [E][T]
  int wv   = threadIdx.x >> 5;
  int lane = threadIdx.x & 31;
  int token = blockIdx.x * (blockDim.x >> 5) + wv;
  if (token >= TT) return;
  size_t base = (size_t)token * DD;

  float xv[32];
  #pragma unroll
  for (int k = 0; k < 32; ++k) xv[k] = x[base + lane + 32 * k];

  float s = 0.f;
  #pragma unroll
  for (int k = 0; k < 32; ++k) s += xv[k];
  float mu = wredsum(s) * (1.0f / DD);

  float v = 0.f;
  #pragma unroll
  for (int k = 0; k < 32; ++k) { float d = xv[k] - mu; v += d * d; }
  float var = wredsum(v) * (1.0f / DD);
  float rs = rsqrtf(var + 1e-5f);

  float xnv[32];
  #pragma unroll
  for (int k = 0; k < 32; ++k) {
    int d = lane + 32 * k;
    xnv[k] = (xv[k] - mu) * rs * gamma[d] + beta[d];
    out[base + d] = xv[k];            // residual
    xnb[base + d] = f2bf(xnv[k]);
  }

  float logit[EE];
  #pragma unroll
  for (int e = 0; e < EE; ++e) {
    float p = 0.f;
    #pragma unroll
    for (int k = 0; k < 32; ++k) p += xnv[k] * Wg[e * DD + lane + 32 * k];
    logit[e] = wredsum(p);
  }

  // softmax over 8 (replicated per lane; lane 0 commits)
  float mx = logit[0];
  #pragma unroll
  for (int e = 1; e < EE; ++e) mx = fmaxf(mx, logit[e]);
  float p[EE]; float ps = 0.f;
  #pragma unroll
  for (int e = 0; e < EE; ++e) { p[e] = __expf(logit[e] - mx); ps += p[e]; }
  float inv = 1.0f / ps;
  #pragma unroll
  for (int e = 0; e < EE; ++e) p[e] *= inv;

  int i1 = 0; float t1 = p[0];
  #pragma unroll
  for (int e = 1; e < EE; ++e) if (p[e] > t1) { t1 = p[e]; i1 = e; }
  int i2 = (i1 == 0) ? 1 : 0; float t2 = p[i2];
  #pragma unroll
  for (int e = 0; e < EE; ++e) if (e != i1 && p[e] > t2) { t2 = p[e]; i2 = e; }

  float denom = t1 + t2 + 1e-9f;
  float c0 = t1 / denom, c1 = t2 / denom;

  if (lane == 0) {
    int pos = atomicAdd(&cnt[i1], 1);
    tokL[i1 * TT + pos] = token; gateL[i1 * TT + pos] = c0;
    pos = atomicAdd(&cnt[i2], 1);
    tokL[i2 * TT + pos] = token; gateL[i2 * TT + pos] = c1;
  }
}

// ---------------------------------------------------------------------------
// Kernel 3: per-expert fused FFN. Block = 512 threads = 16 waves, 32-token
// tile split into two 16-row halves so each weight fragment feeds 2 WMMAs.
// Stage 1: h = gelu(xn * W1t + b1) in 256-wide F chunks; A fragments gathered
//          straight from global (WGP$/L2 absorb the 16x reuse across waves).
// Stage 2: y += h * W2t; each wave owns 64 of the 1024 output columns
//          (4 col-tiles x 2 row-halves = 8 f32 accumulator tiles).
// Epilogue: out[token] += gate * (y + b2) via float atomics (commutative,
//           deterministic: each token gets exactly 2 contributions).
// ---------------------------------------------------------------------------
__global__ void __launch_bounds__(512)
moe_ffn(const u16* __restrict__ xnb,      // [T][D] bf16
        const u16* __restrict__ w1t,      // [E][F][D] bf16
        const u16* __restrict__ w2t,      // [E][D][F] bf16
        const float* __restrict__ pb1,    // [E][F]
        const float* __restrict__ pb2,    // [E][D]
        const int* __restrict__ cnt,
        const int* __restrict__ tokL,
        const float* __restrict__ gateL,
        float* __restrict__ out) {
  const int e = blockIdx.y;
  const int cntE = cnt[e];
  const int t0 = blockIdx.x * 32;
  if (t0 >= cntE) return;   // uniform early exit (before any barrier)

  __shared__ u16 H[32][256 + 8];     // 16.5 KB gelu(h) chunk, 32 token rows
  __shared__ int   sTok[32];
  __shared__ float sGate[32];

  const int tid  = threadIdx.x;
  const int wv   = tid >> 5;             // 0..15
  const int lane = tid & 31;
  const int nIdx = lane & 15;
  const int koff = (lane >> 4) << 3;     // 0 or 8: WMMA 16-bit fragment K-split

  if (tid < 32) {
    int i = t0 + tid;
    bool ok = i < cntE;
    sTok[tid]  = ok ? tokL[e * TT + i] : -1;
    sGate[tid] = ok ? gateL[e * TT + i] : 0.f;
  }
  __syncthreads();

  // per-lane A-row base pointers (dummy row 0 for invalid slots; results of
  // invalid rows are discarded in the epilogue — rows are independent in WMMA)
  int tokLo = sTok[nIdx];        if (tokLo < 0) tokLo = 0;
  int tokHi = sTok[16 + nIdx];   if (tokHi < 0) tokHi = 0;
  const u16* aRow0 = xnb + (size_t)tokLo * DD + koff;   // rows 0..15
  const u16* aRow1 = xnb + (size_t)tokHi * DD + koff;   // rows 16..31

  const u16* W1e = w1t + (size_t)e * FF * DD;
  const u16* W2e = w2t + (size_t)e * DD * FF;

  const v8f vzero = {0.f, 0.f, 0.f, 0.f, 0.f, 0.f, 0.f, 0.f};
  v8f yLo[4], yHi[4];                         // 8 accum tiles = 64 VGPRs
  #pragma unroll
  for (int t = 0; t < 4; ++t) { yLo[t] = vzero; yHi[t] = vzero; }

  for (int fc = 0; fc < FF; fc += 256) {
    // ---- stage 1: h chunk; wave computes 16 cols x 32 rows ----
    v8f hLo = vzero, hHi = vzero;
    const int colBase = fc + wv * 16;
    const u16* bcol = W1e + (size_t)(colBase + nIdx) * DD + koff;
    for (int k0 = 0; k0 < DD; k0 += 32) {
      Frag a0, a1, b;
      a0.u[0] = *(const uint4*)(aRow0 + k0);
      a0.u[1] = *(const uint4*)(aRow0 + k0 + 16);
      a1.u[0] = *(const uint4*)(aRow1 + k0);
      a1.u[1] = *(const uint4*)(aRow1 + k0 + 16);
      b.u[0]  = *(const uint4*)(bcol + k0);
      b.u[1]  = *(const uint4*)(bcol + k0 + 16);
      __builtin_prefetch((const void*)(bcol + k0 + 64), 0, 0);  // global_prefetch_b8
      hLo = __builtin_amdgcn_wmma_f32_16x16x32_bf16(false, a0.v, false, b.v,
                                                    (short)0, hLo, false, false);
      hHi = __builtin_amdgcn_wmma_f32_16x16x32_bf16(false, a1.v, false, b.v,
                                                    (short)0, hHi, false, false);
    }
    // bias + gelu -> LDS (bf16), both row halves
    {
      int hc = wv * 16 + nIdx;
      float bb = pb1[e * FF + colBase + nIdx];
      #pragma unroll
      for (int r = 0; r < 8; ++r) {
        int m = r + koff;   // C-layout row within the 16-row half
        H[m][hc]      = f2bf(gelu_tanh(hLo[r] + bb));
        H[16 + m][hc] = f2bf(gelu_tanh(hHi[r] + bb));
      }
    }
    __syncthreads();

    // ---- stage 2: y += H * W2t-chunk; wave owns cols [wv*64, wv*64+64) ----
    for (int kk = 0; kk < 256; kk += 32) {
      Frag a0, a1;
      const u16* ap0 = &H[nIdx][kk + koff];
      const u16* ap1 = &H[16 + nIdx][kk + koff];
      a0.u[0] = *(const uint4*)ap0;
      a0.u[1] = *(const uint4*)(ap0 + 16);
      a1.u[0] = *(const uint4*)ap1;
      a1.u[1] = *(const uint4*)(ap1 + 16);
      #pragma unroll
      for (int t = 0; t < 4; ++t) {
        int col = wv * 64 + t * 16 + nIdx;
        const u16* bp = W2e + (size_t)col * FF + fc + kk + koff;
        Frag b;
        b.u[0] = *(const uint4*)bp;
        b.u[1] = *(const uint4*)(bp + 16);
        yLo[t] = __builtin_amdgcn_wmma_f32_16x16x32_bf16(false, a0.v, false, b.v,
                                                         (short)0, yLo[t],
                                                         false, false);
        yHi[t] = __builtin_amdgcn_wmma_f32_16x16x32_bf16(false, a1.v, false, b.v,
                                                         (short)0, yHi[t],
                                                         false, false);
      }
    }
    __syncthreads();   // H reused next chunk
  }

  // ---- epilogue: out[token] += gate * (y + b2) ----
  #pragma unroll
  for (int t = 0; t < 4; ++t) {
    int col = wv * 64 + t * 16 + nIdx;
    float bias = pb2[e * DD + col];
    #pragma unroll
    for (int r = 0; r < 8; ++r) {
      int m = r + koff;
      int tok = sTok[m];
      if (tok >= 0) {
        float val = sGate[m] * (yLo[t][r] + bias);
        atomicAdd(&out[(size_t)tok * DD + col], val);
      }
      int tok2 = sTok[16 + m];
      if (tok2 >= 0) {
        float val = sGate[16 + m] * (yHi[t][r] + bias);
        atomicAdd(&out[(size_t)tok2 * DD + col], val);
      }
    }
  }
}

// ---------------------------------------------------------------------------
// Launch
// ---------------------------------------------------------------------------
extern "C" void kernel_launch(void* const* d_in, const int* in_sizes, int n_in,
                              void* d_out, int out_size, void* d_ws, size_t ws_size,
                              hipStream_t stream) {
  const float* x     = (const float*)d_in[0];
  const float* gamma = (const float*)d_in[1];
  const float* beta  = (const float*)d_in[2];
  const float* Wg    = (const float*)d_in[3];
  const float* W1    = (const float*)d_in[4];
  const float* b1    = (const float*)d_in[5];
  const float* W2    = (const float*)d_in[6];
  const float* b2    = (const float*)d_in[7];
  float* out = (float*)d_out;

  // workspace layout
  char* ws = (char*)d_ws;
  size_t off = 0;
  auto take = [&](size_t bytes) { char* p = ws + off; off = (off + bytes + 255) & ~(size_t)255; return p; };
  u16*   xnb   = (u16*)  take((size_t)TT * DD * sizeof(u16));        // 16 MB
  u16*   w1t   = (u16*)  take((size_t)EE * FF * DD * sizeof(u16));   // 64 MB
  u16*   w2t   = (u16*)  take((size_t)EE * DD * FF * sizeof(u16));   // 64 MB
  int*   cnt   = (int*)  take(EE * sizeof(int));
  int*   tokL  = (int*)  take((size_t)EE * TT * sizeof(int));
  float* gateL = (float*)take((size_t)EE * TT * sizeof(float));
  (void)ws_size; (void)in_sizes; (void)n_in; (void)out_size;

  zero_counts<<<1, 32, 0, stream>>>(cnt);

  const long long wtot = (long long)EE * DD * FF;
  convert_transpose<<<16384, 256, 0, stream>>>(W1, w1t, DD, FF, wtot); // [E][F][D]
  convert_transpose<<<16384, 256, 0, stream>>>(W2, w2t, FF, DD, wtot); // [E][D][F]

  ln_gate<<<TT / 8, 256, 0, stream>>>(x, gamma, beta, Wg, out, xnb, cnt, tokL, gateL);

  dim3 grid(TT / 32, EE);
  moe_ffn<<<grid, 512, 0, stream>>>(xnb, w1t, w2t, b1, b2, cnt, tokL, gateL, out);
}
